// RefLocal5_61864708932019
// MI455X (gfx1250) — matrix operand order repaired and checked by reading, not compiled, not run on hardware
//
#include <hip/hip_runtime.h>

// Fused local-window cross-attention for MI455X (gfx1250, wave32, WMMA).
// One workgroup = one 16x16 pixel tile. Everything stays in LDS:
//   stage main tile + 20x20 ref halo + weights (f32 -> f16)
//   WMMA GEMM: cr = refHalo(400x64) x Wr(64x64)   -> LDS f16
//   WMMA GEMM: cm = mainTile(256x64) x Wm(64x64)  -> LDS f16 (aliases sRef arena)
//   VALU: 25 windowed dots per pixel + register softmax -> attn (aliases sMain arena)
//   WMMA GEMM: out = attn(256x32) x Wg(32x64, zero-padded) -> global f32
// LDS arenas are aliased across phases: 155,648 B total -> 2 workgroups per
// 320KB WGP, so one block's HBM staging overlaps another's WMMA/VALU phases.

#define H_IMG 256
#define W_IMG 256
#define BATCH 4
#define CCH   64            // channels (C == D == 64)
#define TS    16            // tile side in pixels
#define HALO  20            // TS + 2*pad
#define NP    (TS * TS)     // 256 pixels per tile
#define NH    (HALO * HALO) // 400 halo pixels
#define NWAVES   8
#define NTHREADS 256

typedef _Float16 v16h __attribute__((ext_vector_type(16)));
typedef float    v8f  __attribute__((ext_vector_type(8)));

// GEMM: D(Mx64,f16) = A(MxKdim,f16, row-major, ld=64) x B(Kdimx64,f16, row-major)
// 16x16x32 tiles via v_wmma_f32_16x16x32_f16; waves strided over output tiles.
__device__ __forceinline__ void gemm_tiles_f16(const _Float16* __restrict__ sA, int M,
                                               const _Float16* __restrict__ sB,
                                               _Float16* __restrict__ sD,
                                               int Kdim, int wid, int lane) {
  const int tiles = (M >> 4) << 2;       // (M/16) row-tiles * 4 col-tiles (N=64)
  const int hf = lane >> 4, lo = lane & 15;
  for (int t = wid; t < tiles; t += NWAVES) {
    const int m0 = (t >> 2) << 4;
    const int n0 = (t & 3) << 4;
    v8f c = {0.f, 0.f, 0.f, 0.f, 0.f, 0.f, 0.f, 0.f};
    for (int k0 = 0; k0 < Kdim; k0 += 32) {
      v16h a, b;
      // A fragment: lane half hf -> K base hf*8; slots 0..7 -> K {0..7}, 8..15 -> K {16..23}
      const _Float16* ar = sA + (m0 + lo) * CCH + k0 + hf * 8;
#pragma unroll
      for (int i = 0; i < 8; ++i) {
        const int kk = (i < 4) ? (2 * i) : (2 * i + 8);
        a[2 * i]     = ar[kk];
        a[2 * i + 1] = ar[kk + 1];
      }
      // B fragment: lane half hf -> K base hf*8; slots 0..7 -> K {0..7}, 8..15 -> K {16..23}; column = lane&15
      const _Float16* br = sB + (size_t)(k0 + hf * 8) * CCH + n0 + lo;
#pragma unroll
      for (int e = 0; e < 16; ++e) b[e] = br[(e < 8 ? e : e + 8) * CCH];
      c = __builtin_amdgcn_wmma_f32_16x16x32_f16(false, a, false, b,
                                                 (short)0, c, false, false);
    }
    // D layout: VGPR r, lanes 0-15 -> M=r, lanes 16-31 -> M=8+r; N = lane&15
#pragma unroll
    for (int r = 0; r < 8; ++r)
      sD[(m0 + hf * 8 + r) * CCH + n0 + lo] = (_Float16)c[r];
  }
}

__global__ __launch_bounds__(NTHREADS)
void local_attn_fused(const float* __restrict__ gmain, const float* __restrict__ gref,
                      const float* __restrict__ gWm, const float* __restrict__ gWr,
                      const float* __restrict__ gWg, float* __restrict__ gout) {
  extern __shared__ _Float16 smem[];
  // Arena layout (in halves), aliased across phases:
  //   [0,     25600) : sRef halo staging   -> sCm (16384) after ref GEMM
  //   [25600, 51200) : sCr
  //   [51200, 67584) : sMain tile staging  -> sAttn (8192) after main GEMM
  //   [67584, 71680) : sWm
  //   [71680, 75776) : sWr
  //   [75776, 77824) : sWg (32x64, rows 25..31 zero)
  _Float16* sRef  = smem;
  _Float16* sCm   = smem;                      // aliases sRef (phase-disjoint)
  _Float16* sCr   = smem + NH * CCH;
  _Float16* sMain = smem + 2 * NH * CCH;
  _Float16* sAttn = smem + 2 * NH * CCH;       // aliases sMain (phase-disjoint)
  _Float16* sWm   = smem + 2 * NH * CCH + NP * CCH;
  _Float16* sWr   = sWm + CCH * CCH;
  _Float16* sWg   = sWr + CCH * CCH;

  const int tid  = threadIdx.x;
  const int wid  = tid >> 5;
  const int lane = tid & 31;
  const int x0 = blockIdx.x * TS;
  const int y0 = blockIdx.y * TS;
  const int bz = blockIdx.z;

  // ---- Stage weights (f32 -> f16) ----
  for (int i = tid; i < (CCH * CCH) / 4; i += NTHREADS) {
    const float4 vm = reinterpret_cast<const float4*>(gWm)[i];
    const float4 vr = reinterpret_cast<const float4*>(gWr)[i];
    _Float16* dm = sWm + i * 4;
    _Float16* dr = sWr + i * 4;
    dm[0] = (_Float16)vm.x; dm[1] = (_Float16)vm.y; dm[2] = (_Float16)vm.z; dm[3] = (_Float16)vm.w;
    dr[0] = (_Float16)vr.x; dr[1] = (_Float16)vr.y; dr[2] = (_Float16)vr.z; dr[3] = (_Float16)vr.w;
  }
  for (int i = tid; i < (32 * CCH) / 4; i += NTHREADS) {
    const int flat = i * 4;
    const int row = flat >> 6, col = flat & 63;
    float4 v{0.f, 0.f, 0.f, 0.f};
    if (row < 25) v = *reinterpret_cast<const float4*>(gWg + row * CCH + col);
    _Float16* d = sWg + flat;
    d[0] = (_Float16)v.x; d[1] = (_Float16)v.y; d[2] = (_Float16)v.z; d[3] = (_Float16)v.w;
  }
  // ---- Stage main tile (f32 -> f16), coalesced float4 ----
  for (int i = tid; i < (NP * CCH) / 4; i += NTHREADS) {
    const int flat = i * 4;
    const int p = flat >> 6, ch = flat & 63;
    const int py = p >> 4, px = p & 15;
    const float4 v = *reinterpret_cast<const float4*>(
        gmain + ((size_t)((bz * H_IMG + y0 + py) * W_IMG + x0 + px)) * CCH + ch);
    _Float16* d = sMain + p * CCH + ch;
    d[0] = (_Float16)v.x; d[1] = (_Float16)v.y; d[2] = (_Float16)v.z; d[3] = (_Float16)v.w;
  }
  // ---- Stage 20x20 ref halo with zero border (== reference's zero-pad, conv is linear) ----
  for (int i = tid; i < (NH * CCH) / 4; i += NTHREADS) {
    const int flat = i * 4;
    const int p = flat >> 6, ch = flat & 63;
    const int hy = p / HALO, hx = p % HALO;
    const int gy = y0 + hy - 2, gx = x0 + hx - 2;
    float4 v{0.f, 0.f, 0.f, 0.f};
    if (gy >= 0 && gy < H_IMG && gx >= 0 && gx < W_IMG)
      v = *reinterpret_cast<const float4*>(
          gref + ((size_t)((bz * H_IMG + gy) * W_IMG + gx)) * CCH + ch);
    _Float16* d = sRef + p * CCH + ch;
    d[0] = (_Float16)v.x; d[1] = (_Float16)v.y; d[2] = (_Float16)v.z; d[3] = (_Float16)v.w;
  }
  __syncthreads();

  // ---- WMMA conv 1: cr = refHalo(400x64) x Wr ----
  gemm_tiles_f16(sRef, NH, sWr, sCr, CCH, wid, lane);
  __syncthreads();   // sRef arena now dead -> reuse as sCm

  // ---- WMMA conv 2: cm = mainTile(256x64) x Wm (D written into sRef arena) ----
  gemm_tiles_f16(sMain, NP, sWm, sCm, CCH, wid, lane);
  __syncthreads();   // sMain arena now dead -> reuse as sAttn

  // ---- Attention: one pixel per thread; 25 dots (len 64), fp32 accumulate, softmax ----
  {
    const int p = tid;
    const int py = p >> 4, px = p & 15;
    float cmv[CCH];
#pragma unroll
    for (int d2 = 0; d2 < CCH; ++d2) cmv[d2] = (float)sCm[p * CCH + d2];
    float lg[25];
#pragma unroll 1
    for (int k = 0; k < 25; ++k) {
      const int di = k / 5, dj = k % 5;
      const _Float16* cr = sCr + ((py + di) * HALO + (px + dj)) * CCH;
      float acc = 0.f;
#pragma unroll
      for (int d2 = 0; d2 < CCH; ++d2) acc += cmv[d2] * (float)cr[d2];
      lg[k] = acc;
    }
    float mx = lg[0];
#pragma unroll
    for (int k = 1; k < 25; ++k) mx = fmaxf(mx, lg[k]);
    float s = 0.f;
#pragma unroll
    for (int k = 0; k < 25; ++k) { const float e = __expf(lg[k] - mx); lg[k] = e; s += e; }
    const float inv = 1.f / s;
#pragma unroll
    for (int k = 0; k < 25; ++k) sAttn[p * 32 + k] = (_Float16)(lg[k] * inv);
#pragma unroll
    for (int k = 25; k < 32; ++k) sAttn[p * 32 + k] = (_Float16)0.f;
  }
  __syncthreads();

  // ---- Final WMMA GEMM: out = attn(256x32) x Wg(32x64), written to global ----
  {
    const int hf = lane >> 4, lo = lane & 15;
    for (int t = wid; t < 64; t += NWAVES) {
      const int m0 = (t >> 2) << 4;
      const int n0 = (t & 3) << 4;
      v16h a, b;
      const _Float16* ar = sAttn + (m0 + lo) * 32 + hf * 8;
#pragma unroll
      for (int i = 0; i < 8; ++i) {
        const int kk = (i < 4) ? (2 * i) : (2 * i + 8);
        a[2 * i]     = ar[kk];
        a[2 * i + 1] = ar[kk + 1];
      }
      const _Float16* br = sWg + (size_t)(hf * 8) * CCH + n0 + lo;
#pragma unroll
      for (int e = 0; e < 16; ++e) b[e] = br[(e < 8 ? e : e + 8) * CCH];
      v8f c = {0.f, 0.f, 0.f, 0.f, 0.f, 0.f, 0.f, 0.f};
      c = __builtin_amdgcn_wmma_f32_16x16x32_f16(false, a, false, b,
                                                 (short)0, c, false, false);
#pragma unroll
      for (int r = 0; r < 8; ++r) {
        const int p = m0 + hf * 8 + r;                  // pixel index in tile
        gout[((size_t)((bz * H_IMG + y0 + (p >> 4)) * W_IMG + x0 + (p & 15))) * CCH
             + n0 + lo] = c[r];
      }
    }
  }
}

extern "C" void kernel_launch(void* const* d_in, const int* in_sizes, int n_in,
                              void* d_out, int out_size, void* d_ws, size_t ws_size,
                              hipStream_t stream) {
  (void)in_sizes; (void)n_in; (void)out_size; (void)d_ws; (void)ws_size;
  const float* gmain = (const float*)d_in[0];
  const float* gref  = (const float*)d_in[1];
  const float* gWm   = (const float*)d_in[2];
  const float* gWr   = (const float*)d_in[3];
  const float* gWg   = (const float*)d_in[4];
  float* gout = (float*)d_out;

  dim3 grid(W_IMG / TS, H_IMG / TS, BATCH);   // 16 x 16 x 4 = 1024 workgroups
  dim3 block(NTHREADS);                       // 8 wave32s
  const size_t smem_halves = (size_t)2 * NH * CCH   // sRef(+sCm alias) + sCr
                           + (size_t)NP * CCH       // sMain (+sAttn alias)
                           + (size_t)CCH * CCH * 2  // sWm + sWr
                           + (size_t)32 * CCH;      // sWg (padded)
  const size_t smem_bytes = smem_halves * sizeof(_Float16); // 155,648 B -> 2 WGs/WGP
  local_attn_fused<<<grid, block, smem_bytes, stream>>>(gmain, gref, gWm, gWr, gWg, gout);
}
